// MultiHeadAttention_34050500723440
// MI455X (gfx1250) — matrix operand order repaired and checked by reading, not compiled, not run on hardware
//
#include <hip/hip_runtime.h>

// ---------------------------------------------------------------------------
// MHA forward for MI455X (gfx1250, wave32, WMMA bf16).
// B=2, S=2048, HID=2048, NH=16, HD=128.
// ---------------------------------------------------------------------------

typedef unsigned short u16;
typedef __attribute__((ext_vector_type(16))) __bf16          v16bf;
typedef __attribute__((ext_vector_type(8)))  float           v8f;
typedef __attribute__((ext_vector_type(16))) unsigned short  v16u;
typedef __attribute__((ext_vector_type(8)))  unsigned short  v8u;

#define WMMA_BF16(a, b, c) \
  __builtin_amdgcn_wmma_f32_16x16x32_bf16(false, (a), false, (b), (short)0, (c), false, false)

static __device__ __forceinline__ u16 f32_to_bf16(float f) {
  unsigned u = __builtin_bit_cast(unsigned, f);
  unsigned r = (u + 0x7FFFu + ((u >> 16) & 1u)) >> 16;
  return (u16)r;
}
static __device__ __forceinline__ float bf16_to_f32(u16 h) {
  unsigned u = ((unsigned)h) << 16;
  return __builtin_bit_cast(float, u);
}
static __device__ __forceinline__ v16bf cat_bf16(v8u lo, v8u hi) {
  v16u v = __builtin_shufflevector(lo, hi, 0,1,2,3,4,5,6,7,8,9,10,11,12,13,14,15);
  return __builtin_bit_cast(v16bf, v);
}
static __device__ __forceinline__ v16bf as_bf16(v16u v) {
  return __builtin_bit_cast(v16bf, v);
}
static __device__ __forceinline__ void wave_sched_barrier() {
#if __has_builtin(__builtin_amdgcn_wave_barrier)
  __builtin_amdgcn_wave_barrier();
#else
  asm volatile("" ::: "memory");
#endif
}

// ---------------------------------------------------------------------------
// f32 -> bf16 conversion (grid-stride)
// ---------------------------------------------------------------------------
__global__ void cvt_f32_bf16(const float* __restrict__ in, u16* __restrict__ out, size_t n) {
  size_t i = (size_t)blockIdx.x * blockDim.x + threadIdx.x;
  size_t stride = (size_t)gridDim.x * blockDim.x;
  for (; i < n; i += stride) out[i] = f32_to_bf16(in[i]);
}

// ---------------------------------------------------------------------------
// GEMM: C[M,N] = A[M,K] * Bw[N,K]^T,  A/Bw bf16 row-major, acc f32.
// One wave -> 32(M) x 64(N) tile, K loop step 32, 8 wmma / iter
// (each B fragment reused by 2 A rows-of-16; each A fragment reused 4x).
// ---------------------------------------------------------------------------
template <bool OUT_BF16>
__global__ __launch_bounds__(256) void gemm_bf16_wmma(
    const u16* __restrict__ A, const u16* __restrict__ Bw, void* __restrict__ Cout,
    int M, int N, int K) {
  const int lane = threadIdx.x & 31;
  const int wave = threadIdx.x >> 5;
  const int half = lane >> 4;
  const int l15  = lane & 15;
  const int tiles_n = N >> 6;
  const int t  = blockIdx.x * 8 + wave;
  const int tm = t / tiles_n;
  const int tn = t - tm * tiles_n;
  if (tm * 32 >= M) return;

  // A fragments: two 16-row groups; per-lane K offset per the 16-bit A layout.
  const u16* ap0 = A + (size_t)(tm * 32 + l15) * K + (half << 3);
  const u16* ap1 = ap0 + (size_t)16 * K;
  // B fragments: 4 column groups of 16; lane owns weight row (tn*64+j*16+l15).
  const u16* bp  = Bw + (size_t)(tn * 64 + l15) * K + (half << 3);

  v8f acc[2][4] = {};
  for (int k0 = 0; k0 < K; k0 += 32) {
    __builtin_prefetch(ap0 + k0 + 128, 0, 0);  // global_prefetch_b8: next A chunks
    __builtin_prefetch(ap1 + k0 + 128, 0, 0);
    v16bf a0 = cat_bf16(*(const v8u*)(ap0 + k0), *(const v8u*)(ap0 + k0 + 16));
    v16bf a1 = cat_bf16(*(const v8u*)(ap1 + k0), *(const v8u*)(ap1 + k0 + 16));
#pragma unroll
    for (int j = 0; j < 4; ++j) {
      v16bf b = cat_bf16(*(const v8u*)(bp + (size_t)j * 16 * K + k0), *(const v8u*)(bp + (size_t)j * 16 * K + k0 + 16));
      acc[0][j] = WMMA_BF16(a0, b, acc[0][j]);
      acc[1][j] = WMMA_BF16(a1, b, acc[1][j]);
    }
  }

  const int col0 = tn * 64 + l15;
#pragma unroll
  for (int i = 0; i < 2; ++i)
#pragma unroll
    for (int j = 0; j < 4; ++j)
#pragma unroll
      for (int v = 0; v < 8; ++v) {
        const int r = tm * 32 + i * 16 + v + 8 * half;  // C layout: row = vgpr + 8*half
        const int c = col0 + j * 16;                    //           col = lane & 15
        if (OUT_BF16) ((u16*)Cout)[(size_t)r * N + c] = f32_to_bf16(acc[i][j][v]);
        else          ((float*)Cout)[(size_t)r * N + c] = acc[i][j][v];
      }
}

// ---------------------------------------------------------------------------
// RoPE on q,k + head transpose. qkv: [B,S,3,NH,HD] bf16.
// Writes Q,K: [B,NH,S,HD] bf16 ; V transposed: [B,NH,HD,S] bf16.
// ---------------------------------------------------------------------------
__global__ __launch_bounds__(256) void rope_transpose(
    const u16* __restrict__ qkv, const float* __restrict__ cosb, const float* __restrict__ sinb,
    u16* __restrict__ Q, u16* __restrict__ Kb, u16* __restrict__ VT) {
  size_t idx = (size_t)blockIdx.x * blockDim.x + threadIdx.x;  // B*S*NH*HD threads
  const int d = (int)(idx & 127);
  size_t t = idx >> 7;
  const int h = (int)(t & 15);  t >>= 4;
  const int s = (int)(t & 2047); t >>= 11;
  const int b = (int)t;

  const u16* base = qkv + ((size_t)(b * 2048 + s)) * 6144;  // 3*HID per token
  const int hd = h * 128 + d;
  const float qv = bf16_to_f32(base[hd]);
  const float kv = bf16_to_f32(base[2048 + hd]);
  const float vv = bf16_to_f32(base[4096 + hd]);

  const int   d2  = (d < 64) ? d + 64 : d - 64;
  const float sgn = (d < 64) ? -1.0f : 1.0f;
  const float qr = bf16_to_f32(base[h * 128 + d2]);
  const float kr = bf16_to_f32(base[2048 + h * 128 + d2]);
  const float c  = cosb[s * 128 + d];
  const float sn = sinb[s * 128 + d];

  const size_t off  = (((size_t)(b * 16 + h)) * 2048 + s) * 128 + d;  // [B,NH,S,HD]
  const size_t offT = (((size_t)(b * 16 + h)) * 128 + d) * 2048 + s;  // [B,NH,HD,S]
  Q[off]  = f32_to_bf16(qv * c + sgn * qr * sn);
  Kb[off] = f32_to_bf16(kv * c + sgn * kr * sn);
  VT[offT] = f32_to_bf16(vv);
}

// ---------------------------------------------------------------------------
// Flash attention (causal). One wave owns a 16-query tile of one (b,h).
// Q,K: [B,NH,S,HD] bf16 ; VT: [B,NH,HD,S] bf16 ; ctx out: [B,S,HID] bf16.
// ---------------------------------------------------------------------------
__global__ __launch_bounds__(128) void flash_attn(
    const u16* __restrict__ Q, const u16* __restrict__ Kb, const u16* __restrict__ VT,
    u16* __restrict__ ctx) {
  __shared__ u16 lds[4][16 * 32];  // wave-private 16x32 bf16 P tile

  const int lane = threadIdx.x & 31;
  const int wave = threadIdx.x >> 5;
  const int half = lane >> 4;
  const int l15  = lane & 15;

  const int gw = blockIdx.x * 4 + wave;   // 4096 waves total
  const int qt = gw & 127;                // q tile within (b,h), S/16 = 128
  const int bh = gw >> 7;                 // 0..31
  const int q0 = qt * 16;

  const u16* qbase = Q  + ((size_t)bh * 2048 + q0) * 128;
  const u16* kbase = Kb + (size_t)bh * 2048 * 128;
  const u16* vbase = VT + (size_t)bh * 128 * 2048;

  // Load the 16x128 Q tile as 4 A-fragments (kept in registers the whole time).
  v16bf qa[4];
  {
    const u16* qp = qbase + (size_t)l15 * 128 + (half << 4);
#pragma unroll
    for (int i = 0; i < 4; ++i) {
      v8u lo = *(const v8u*)(qp + i * 32);
      v8u hi = *(const v8u*)(qp + i * 32 + 8);
      qa[i] = cat_bf16(lo, hi);
    }
  }

  v8f o[8] = {};              // 16x128 f32 output accumulator (8 C-frags)
  float m[8], l[8];
#pragma unroll
  for (int v = 0; v < 8; ++v) { m[v] = -3.0e38f; l[v] = 0.0f; }

  const float scale = 0.088388347648318447f;  // 1/sqrt(128)
  u16* pl = &lds[wave][0];

  for (int kb0 = 0; kb0 < q0 + 16; kb0 += 32) {
    // ---- scores: two 16-key sub-blocks, K-dim = HD = 128 (4 wmma each) ----
    v8f s0 = {}, s1 = {};
#pragma unroll
    for (int i = 0; i < 4; ++i) {
      const u16* kp0 = kbase + (size_t)(kb0 + l15) * 128 + i * 32 + (half << 4);
      s0 = WMMA_BF16(qa[i], as_bf16(*(const v16u*)kp0), s0);
      s1 = WMMA_BF16(qa[i], as_bf16(*(const v16u*)(kp0 + 16 * 128)), s1);
    }

    // ---- scale + causal mask + online softmax ----
    float alpha[8];
#pragma unroll
    for (int v = 0; v < 8; ++v) {
      const int r  = q0 + v + 8 * half;    // global query row for this element
      const int c0 = kb0 + l15;            // global key col (sub-block 0)
      float x0 = s0[v] * scale; if (c0 > r)      x0 = -1.0e9f;
      float x1 = s1[v] * scale; if (c0 + 16 > r) x1 = -1.0e9f;

      float x = fmaxf(x0, x1);             // row max across 16 lanes of the half
#pragma unroll
      for (int off = 1; off < 16; off <<= 1) x = fmaxf(x, __shfl_xor(x, off, 32));
      const float mn = fmaxf(m[v], x);
      const float a  = __expf(m[v] - mn);
      const float p0 = __expf(x0 - mn);
      const float p1 = __expf(x1 - mn);
      s0[v] = p0; s1[v] = p1;
      float rs = p0 + p1;
#pragma unroll
      for (int off = 1; off < 16; off <<= 1) rs += __shfl_xor(rs, off, 32);
      l[v] = l[v] * a + rs;
      m[v] = mn;
      alpha[v] = a;
    }

    // rescale running output
#pragma unroll
    for (int j = 0; j < 8; ++j)
#pragma unroll
      for (int v = 0; v < 8; ++v) o[j][v] *= alpha[v];

    // ---- C-layout P -> A-layout via wave-private LDS tile ----
#pragma unroll
    for (int v = 0; v < 8; ++v) {
      const int r = v + 8 * half;
      pl[r * 32 + l15]      = f32_to_bf16(s0[v]);
      pl[r * 32 + 16 + l15] = f32_to_bf16(s1[v]);
    }
    wave_sched_barrier();   // wave-private tile: per-wave LDS ordering suffices
    const u16* pp = pl + l15 * 32 + (half << 4);
    v16bf pa = cat_bf16(*(const v8u*)pp, *(const v8u*)(pp + 8));
    wave_sched_barrier();

    // ---- O += P (16x32) x V (32x16 per d-group), 8 wmma ----
#pragma unroll
    for (int j = 0; j < 8; ++j) {
      const u16* vp = vbase + (size_t)(j * 16 + l15) * 2048 + kb0 + (half << 4);
      o[j] = WMMA_BF16(pa, as_bf16(*(const v16u*)vp), o[j]);
    }
  }

  // ---- normalize + store ctx[b, q, h*128 + d] as bf16 ----
#pragma unroll
  for (int v = 0; v < 8; ++v) l[v] = 1.0f / l[v];
  const int b = bh >> 4, h = bh & 15;
#pragma unroll
  for (int j = 0; j < 8; ++j)
#pragma unroll
    for (int v = 0; v < 8; ++v) {
      const int r = q0 + v + 8 * half;
      const int c = h * 128 + j * 16 + l15;
      ctx[((size_t)b * 2048 + r) * 2048 + c] = f32_to_bf16(o[j][v] * l[v]);
    }
}

// ---------------------------------------------------------------------------
// Host-side orchestration
// ---------------------------------------------------------------------------
extern "C" void kernel_launch(void* const* d_in, const int* in_sizes, int n_in,
                              void* d_out, int out_size, void* d_ws, size_t ws_size,
                              hipStream_t stream) {
  (void)in_sizes; (void)n_in; (void)out_size; (void)ws_size;
  const float* x     = (const float*)d_in[0];   // [2,2048,2048]
  const float* w_qkv = (const float*)d_in[1];   // [6144,2048]
  const float* w_o   = (const float*)d_in[2];   // [2048,2048]
  const float* cosb  = (const float*)d_in[3];   // [2048,128]
  const float* sinb  = (const float*)d_in[4];   // [2048,128]
  float* out = (float*)d_out;                   // [2,2048,2048] f32

  // workspace layout (bf16 elements)
  u16* xb    = (u16*)d_ws;                      //  4096*2048
  u16* wqkvb = xb    + (size_t)4096 * 2048;     //  6144*2048
  u16* wob   = wqkvb + (size_t)6144 * 2048;     //  2048*2048
  u16* qkvb  = wob   + (size_t)2048 * 2048;     //  4096*6144
  u16* qb    = qkvb  + (size_t)4096 * 6144;     //  [B,NH,S,HD]
  u16* kb    = qb    + (size_t)8388608;
  u16* vtb   = kb    + (size_t)8388608;         //  [B,NH,HD,S]
  u16* ctxb  = vtb   + (size_t)8388608;         //  [B,S,HID]

  cvt_f32_bf16<<<4096, 256, 0, stream>>>(x,     xb,    (size_t)4096 * 2048);
  cvt_f32_bf16<<<4096, 256, 0, stream>>>(w_qkv, wqkvb, (size_t)6144 * 2048);
  cvt_f32_bf16<<<4096, 256, 0, stream>>>(w_o,   wob,   (size_t)2048 * 2048);

  // QKV projection: [4096,2048] x [6144,2048]^T -> [4096,6144] bf16
  // tiles: (4096/32) * (6144/64) = 128 * 96 = 12288 waves -> 1536 blocks
  gemm_bf16_wmma<true><<<1536, 256, 0, stream>>>(xb, wqkvb, qkvb, 4096, 6144, 2048);

  // RoPE + head transpose (8.4M threads)
  rope_transpose<<<32768, 256, 0, stream>>>(qkvb, cosb, sinb, qb, kb, vtb);

  // Causal flash attention: 4096 waves, 4 per block
  flash_attn<<<1024, 128, 0, stream>>>(qb, kb, vtb, ctxb);

  // Output projection: [4096,2048] x [2048,2048]^T -> f32 out
  // tiles: 128 * 32 = 4096 waves -> 512 blocks
  gemm_bf16_wmma<false><<<512, 256, 0, stream>>>(ctxb, wob, out, 4096, 2048, 2048);
}
